// ExpertCluster_54288386622044
// MI455X (gfx1250) — matrix operand order repaired and checked by reading, not compiled, not run on hardware
//
#include <hip/hip_runtime.h>
#include <hip/hip_bf16.h>
#include <math.h>

// ---------------------------------------------------------------------------
// MoE expert cluster for MI455X (gfx1250, wave32, WMMA + async global->LDS).
// Compute-bound (1.3 TFLOP vs ~17us HBM weight traffic) -> f16 WMMA
// (v_wmma_f32_16x16x32_f16, f32 accumulate) with double-buffered
// global_load_async_to_lds_b128 staging (ASYNCcnt / s_wait_asynccnt).
// ---------------------------------------------------------------------------

typedef __attribute__((ext_vector_type(16))) _Float16 v16h;
typedef __attribute__((ext_vector_type(8)))  float    v8f;

constexpr int E_  = 8;
constexpr int D_  = 1024;
constexpr int DFF = 4096;
constexpr int B_  = 4;
constexpr int S_  = 2048;
constexpr int KT  = 2;                 // top-k
constexpr int N_  = B_ * S_;           // 8192 tokens
constexpr float LN_EPS = 1e-5f;

constexpr int BM = 128, BN = 128, BK = 32;
constexpr int AS = 40;                 // LDS row stride in halves: 80B rows ->
                                       // 16B-aligned async dests, conflict-free banks
constexpr int TILE_H = BM * AS;        // halves per LDS tile (5120 -> 10240B)

// ---------------------------------------------------------------------------
// WMMA fragment loads (layouts per CDNA5 ISA 7.12.2, 16-bit wave32):
//  A 16x32 (MxK) in LDS [m][k]: lanes 0-15 row M=lane, VGPR v holds
//     K {2v,2v+1} (v<4) / {16+2(v-4),17+..} (v>=4); lanes 16-31 +8 / +24.
//  B 32x16 (KxN) in LDS [n][k] (transposed): lane col n=lane&15,
//     same K mapping as A: VGPR v holds K {2v,2v+1} (v<4) / {16+2(v-4),..}; lanes 16-31 +8.
// ---------------------------------------------------------------------------
__device__ inline v16h load_a_frag(const _Float16* sA, int m, int lane) {
  union { v16h v; uint32_t u[8]; } r;
  const int lhalf = lane >> 4;
#pragma unroll
  for (int vv = 0; vv < 8; ++vv) {
    int kbase = (vv < 4) ? (lhalf ? 8 : 0) : (lhalf ? 24 : 16);
    int k = kbase + 2 * (vv & 3);
    r.u[vv] = *(const uint32_t*)(sA + m * AS + k);
  }
  return r.v;
}

__device__ inline v16h load_b_frag(const _Float16* sB, int n, int lane) {
  union { v16h v; uint32_t u[8]; } r;
  const int kbase = (lane >> 4) ? 8 : 0;
#pragma unroll
  for (int vv = 0; vv < 8; ++vv)
    r.u[vv] = *(const uint32_t*)(sB + n * AS + kbase + ((vv < 4) ? 0 : 16) + 2 * (vv & 3));
  return r.v;
}

__device__ inline v8f wmma_f16(v16h a, v16h b, v8f c) {
  return __builtin_amdgcn_wmma_f32_16x16x32_f16(false, a, false, b,
                                                (short)0, c, false, false);
}

__device__ inline uint32_t lds_lo(const void* p) {
  return (uint32_t)(uintptr_t)p;       // low 32 bits of generic ptr == LDS offset
}

// Async-stage one 128-row x 64-byte tile (f16, contiguous rows of 32 halves)
// into LDS rows of stride 80B.  2 wave-instructions per wave (ASYNCcnt += 2).
__device__ inline void stage_async(const _Float16* g, size_t ldh,
                                   uint32_t lds, int tid) {
#pragma unroll
  for (int i = 0; i < 2; ++i) {
    int c  = tid + i * 256;            // 512 chunks of 16B
    int r  = c >> 2;
    int cc = (c & 3) * 16;             // byte offset within 64B row segment
    const char* ga = (const char*)(g + (size_t)r * ldh) + cc;
    uint32_t    la = lds + (uint32_t)(r * (AS * 2) + cc);
    asm volatile("global_load_async_to_lds_b128 %0, %1, off"
                 :: "v"(la), "v"(ga) : "memory");
  }
}

// ---------------------------------------------------------------------------
// GEMM 1(+2): H = act( XN @ W1t^T [, XN @ W2t^T] )   [N_ x DFF], f16 out
// XN = LN1-applied tokens (f16), W1t/W2t = transposed f16 weights [DFF x D_].
// ACT: 0 = swiglu (silu(h2)*h2*h1), 1 = gelu(h1), 2 = relu(h1)
// ---------------------------------------------------------------------------
template <int ACT>
__global__ __launch_bounds__(256)
void moe_gemm12(const _Float16* __restrict__ xn,
                const _Float16* __restrict__ W1t,
                const _Float16* __restrict__ W2t,
                _Float16* __restrict__ H) {
  constexpr int NTILE = (ACT == 0) ? 6 : 4;    // LDS tiles (double buffered)
  constexpr int OPS   = (ACT == 0) ? 6 : 4;    // async instrs per wave per k-tile
  __shared__ _Float16 smem[NTILE * TILE_H];
  // Buffer pointers computed per use (no constant-initialized LDS ptr arrays:
  // those lower to addrspacecast static initializers that ld.lld rejects).
  auto tileA  = [&](int q) -> _Float16* { return smem + q * TILE_H; };
  auto tileB1 = [&](int q) -> _Float16* { return smem + (2 + q) * TILE_H; };
  auto tileB2 = [&](int q) -> _Float16* {
    return smem + (((ACT == 0) ? 4 : 2) + q) * TILE_H;  // aliases B1 if unused
  };

  const int tid  = threadIdx.x;
  const int lane = tid & 31;
  const int wave = tid >> 5;
  const int wm = wave >> 2;            // 0..1  (64 rows each)
  const int wn = wave & 3;             // 0..3  (32 cols each)
  const int row0 = blockIdx.y * BM;
  const int col0 = blockIdx.x * BN;

  const v8f vz = {0.f,0.f,0.f,0.f,0.f,0.f,0.f,0.f};
  v8f acc1[4][2], acc2[4][2];
#pragma unroll
  for (int i = 0; i < 4; ++i)
#pragma unroll
    for (int j = 0; j < 2; ++j) { acc1[i][j] = vz; acc2[i][j] = vz; }

  auto stage_tile = [&](int tt, int q) {
    int kb = tt * BK;
    stage_async(xn  + (size_t)row0 * D_ + kb, D_, lds_lo(tileA(q)),  tid);
    stage_async(W1t + (size_t)col0 * D_ + kb, D_, lds_lo(tileB1(q)), tid);
    if (ACT == 0)
      stage_async(W2t + (size_t)col0 * D_ + kb, D_, lds_lo(tileB2(q)), tid);
  };

  constexpr int T = D_ / BK;           // 32 k-tiles
  stage_tile(0, 0);
  for (int t = 0; t < T; ++t) {
    const int cur = t & 1;
    if (t + 1 < T) {
      stage_tile(t + 1, 1 - cur);
      asm volatile("s_wait_asynccnt %0" :: "i"(OPS) : "memory");
    } else {
      asm volatile("s_wait_asynccnt 0x0" ::: "memory");
    }
    __syncthreads();                   // tile t visible to whole WG

    const _Float16* pA  = tileA(cur);
    const _Float16* pB1 = tileB1(cur);
    const _Float16* pB2 = tileB2(cur);
    v16h a[4];
#pragma unroll
    for (int i = 0; i < 4; ++i)
      a[i] = load_a_frag(pA, wm * 64 + i * 16 + (lane & 15), lane);
    v16h b1[2], b2[2];
#pragma unroll
    for (int j = 0; j < 2; ++j) {
      b1[j] = load_b_frag(pB1, wn * 32 + j * 16 + (lane & 15), lane);
      if (ACT == 0) b2[j] = load_b_frag(pB2, wn * 32 + j * 16 + (lane & 15), lane);
    }
#pragma unroll
    for (int i = 0; i < 4; ++i)
#pragma unroll
      for (int j = 0; j < 2; ++j) {
        acc1[i][j] = wmma_f16(a[i], b1[j], acc1[i][j]);
        if (ACT == 0) acc2[i][j] = wmma_f16(a[i], b2[j], acc2[i][j]);
      }
    __syncthreads();                   // all reads of buf done before refill
  }

  // --- epilogue: activation, f16 store.  C/D layout: VGPR r -> M = r (+8 hi) ---
  const int lh = (lane >> 4) * 8;
#pragma unroll
  for (int i = 0; i < 4; ++i)
#pragma unroll
    for (int j = 0; j < 2; ++j) {
      int gn = col0 + wn * 32 + j * 16 + (lane & 15);
#pragma unroll
      for (int r = 0; r < 8; ++r) {
        int gm = row0 + wm * 64 + i * 16 + lh + r;
        float h1 = acc1[i][j][r];
        float val;
        if (ACT == 0) {
          float h2 = acc2[i][j][r];
          float sg = 1.0f / (1.0f + __expf(-h2));  // sigmoid
          val = (h2 * sg) * h2 * h1;               // silu(h2)*h2*h1
        } else if (ACT == 1) {
          val = 0.5f * h1 * (1.0f + erff(h1 * 0.70710678118654752f));
        } else {
          val = fmaxf(h1, 0.0f);
        }
        H[(size_t)gm * DFF + gn] = (_Float16)val;
      }
    }
}

// ---------------------------------------------------------------------------
// GEMM 3: Y = x + H @ W3t^T    [N_ x D_], fp32 out.  W3t = [D_ x DFF] f16.
// ---------------------------------------------------------------------------
__global__ __launch_bounds__(256)
void moe_gemm3(const _Float16* __restrict__ H, const _Float16* __restrict__ W3t,
               const float* __restrict__ x, float* __restrict__ Y) {
  constexpr int OPS = 4;
  __shared__ _Float16 smem[4 * TILE_H];
  auto tileA = [&](int q) -> _Float16* { return smem + q * TILE_H; };
  auto tileB = [&](int q) -> _Float16* { return smem + (2 + q) * TILE_H; };

  const int tid  = threadIdx.x;
  const int lane = tid & 31;
  const int wave = tid >> 5;
  const int wm = wave >> 2, wn = wave & 3;
  const int row0 = blockIdx.y * BM;
  const int col0 = blockIdx.x * BN;

  const v8f vz = {0.f,0.f,0.f,0.f,0.f,0.f,0.f,0.f};
  v8f acc[4][2];
#pragma unroll
  for (int i = 0; i < 4; ++i)
#pragma unroll
    for (int j = 0; j < 2; ++j) acc[i][j] = vz;

  auto stage_tile = [&](int tt, int q) {
    int kb = tt * BK;
    stage_async(H   + (size_t)row0 * DFF + kb, DFF, lds_lo(tileA(q)), tid);
    stage_async(W3t + (size_t)col0 * DFF + kb, DFF, lds_lo(tileB(q)), tid);
  };

  constexpr int T = DFF / BK;          // 128 k-tiles
  stage_tile(0, 0);
  for (int t = 0; t < T; ++t) {
    const int cur = t & 1;
    if (t + 1 < T) {
      stage_tile(t + 1, 1 - cur);
      asm volatile("s_wait_asynccnt %0" :: "i"(OPS) : "memory");
    } else {
      asm volatile("s_wait_asynccnt 0x0" ::: "memory");
    }
    __syncthreads();

    const _Float16* pA = tileA(cur);
    const _Float16* pB = tileB(cur);
    v16h a[4];
#pragma unroll
    for (int i = 0; i < 4; ++i)
      a[i] = load_a_frag(pA, wm * 64 + i * 16 + (lane & 15), lane);
    v16h b[2];
#pragma unroll
    for (int j = 0; j < 2; ++j)
      b[j] = load_b_frag(pB, wn * 32 + j * 16 + (lane & 15), lane);
#pragma unroll
    for (int i = 0; i < 4; ++i)
#pragma unroll
      for (int j = 0; j < 2; ++j)
        acc[i][j] = wmma_f16(a[i], b[j], acc[i][j]);
    __syncthreads();
  }

  const int lh = (lane >> 4) * 8;
#pragma unroll
  for (int i = 0; i < 4; ++i)
#pragma unroll
    for (int j = 0; j < 2; ++j) {
      int gn = col0 + wn * 32 + j * 16 + (lane & 15);
#pragma unroll
      for (int r = 0; r < 8; ++r) {
        int gm = row0 + wm * 64 + i * 16 + lh + r;
        size_t gi = (size_t)gm * D_ + gn;
        Y[gi] = x[gi] + acc[i][j][r];          // residual
      }
    }
}

// ---------------------------------------------------------------------------
// Per-expert prep: xn = xhat*ln1w + ln1b (f16), transposed f16 weights.
// ---------------------------------------------------------------------------
__global__ __launch_bounds__(256)
void ln1_apply(const _Float16* __restrict__ xhat, const float* __restrict__ w,
               const float* __restrict__ b, _Float16* __restrict__ xn) {
  int i = blockIdx.x * 256 + threadIdx.x;      // over N_*D_
  int d = i & (D_ - 1);
  xn[i] = (_Float16)((float)xhat[i] * w[d] + b[d]);
}

// W [R x C] fp32 -> WT [C x R] f16 (tiled transpose, coalesced both ways)
__global__ __launch_bounds__(256)
void transpose_to_f16(const float* __restrict__ W, _Float16* __restrict__ WT,
                      int R, int C) {
  __shared__ float t[32][33];
  const int bx = blockIdx.x * 32;              // C offset
  const int by = blockIdx.y * 32;              // R offset
  const int tx = threadIdx.x & 31;
  const int ty = threadIdx.x >> 5;             // 0..7
#pragma unroll
  for (int i = 0; i < 32; i += 8)
    t[ty + i][tx] = W[(size_t)(by + ty + i) * C + bx + tx];
  __syncthreads();
#pragma unroll
  for (int i = 0; i < 32; i += 8)
    WT[(size_t)(bx + ty + i) * R + by + tx] = (_Float16)t[tx][ty + i];
}

// ---------------------------------------------------------------------------
// Row-wise helpers (one block per token, 256 threads)
// ---------------------------------------------------------------------------
__device__ inline void block_reduce2(float& s1, float& s2) {
  __shared__ float r1[256], r2[256];
  int t = threadIdx.x;
  r1[t] = s1; r2[t] = s2;
  __syncthreads();
  for (int o = 128; o > 0; o >>= 1) {
    if (t < o) { r1[t] += r1[t + o]; r2[t] += r2[t + o]; }
    __syncthreads();
  }
  s1 = r1[0]; s2 = r2[0];
}

// LN stats of x -> xhat f16; capacity-penalized top-k softmax -> coef[N_,E_]
__global__ __launch_bounds__(256)
void prep_tokens(const float* __restrict__ x, const float* __restrict__ ew,
                 const int* __restrict__ eidx, const float* __restrict__ capu,
                 const float* __restrict__ cap, _Float16* __restrict__ xhat,
                 float* __restrict__ coef) {
  int n = blockIdx.x;
  const float* row = x + (size_t)n * D_;
  float s = 0.f, s2 = 0.f;
  for (int d = threadIdx.x; d < D_; d += 256) { float v = row[d]; s += v; s2 += v * v; }
  block_reduce2(s, s2);
  float m    = s / D_;
  float var  = s2 / D_ - m * m;
  float rstd = rsqrtf(var + LN_EPS);
  for (int d = threadIdx.x; d < D_; d += 256)
    xhat[(size_t)n * D_ + d] = (_Float16)((row[d] - m) * rstd);
  if (threadIdx.x == 0) {
    float p[KT]; int id[KT]; float mx = -1e30f;
#pragma unroll
    for (int k = 0; k < KT; ++k) {
      id[k] = eidx[n * KT + k];
      float pen = capu[id[k]] / (cap[id[k]] + 1e-8f);
      pen = fminf(fmaxf(pen, 0.f), 2.f);
      p[k] = ew[n * KT + k] / (1.f + pen);
      mx = fmaxf(mx, p[k]);
    }
    float sum = 0.f;
#pragma unroll
    for (int k = 0; k < KT; ++k) { p[k] = __expf(p[k] - mx); sum += p[k]; }
    float c[E_];
#pragma unroll
    for (int e = 0; e < E_; ++e) c[e] = 0.f;
#pragma unroll
    for (int k = 0; k < KT; ++k) c[id[k]] += p[k] / sum;
#pragma unroll
    for (int e = 0; e < E_; ++e) coef[(size_t)n * E_ + e] = c[e];
  }
}

// combined += coef[:,e] * ( LN(Y)*ln2w + ln2b )
__global__ __launch_bounds__(256)
void ln2_combine(const float* __restrict__ Y, const float* __restrict__ w,
                 const float* __restrict__ b, const float* __restrict__ coef,
                 int e, float* __restrict__ combined) {
  int n = blockIdx.x;
  const float* row = Y + (size_t)n * D_;
  float s = 0.f, s2 = 0.f;
  for (int d = threadIdx.x; d < D_; d += 256) { float v = row[d]; s += v; s2 += v * v; }
  block_reduce2(s, s2);
  float m    = s / D_;
  float var  = s2 / D_ - m * m;
  float rstd = rsqrtf(var + LN_EPS);
  float c = coef[(size_t)n * E_ + e];
  for (int d = threadIdx.x; d < D_; d += 256) {
    float eo = (row[d] - m) * rstd * w[d] + b[d];
    combined[(size_t)n * D_ + d] += c * eo;
  }
}

// out = LN(combined)*clw + clb
__global__ __launch_bounds__(256)
void final_ln(const float* __restrict__ cmb, const float* __restrict__ w,
              const float* __restrict__ b, float* __restrict__ out) {
  int n = blockIdx.x;
  const float* row = cmb + (size_t)n * D_;
  float s = 0.f, s2 = 0.f;
  for (int d = threadIdx.x; d < D_; d += 256) { float v = row[d]; s += v; s2 += v * v; }
  block_reduce2(s, s2);
  float m    = s / D_;
  float var  = s2 / D_ - m * m;
  float rstd = rsqrtf(var + LN_EPS);
  for (int d = threadIdx.x; d < D_; d += 256)
    out[(size_t)n * D_ + d] = (row[d] - m) * rstd * w[d] + b[d];
}

// ---------------------------------------------------------------------------
extern "C" void kernel_launch(void* const* d_in, const int* in_sizes, int n_in,
                              void* d_out, int out_size, void* d_ws, size_t ws_size,
                              hipStream_t stream) {
  const float* x    = (const float*)d_in[0];
  const float* ew   = (const float*)d_in[1];
  const int*   eidx = (const int*)  d_in[2];
  const float* ln1w = (const float*)d_in[3];
  const float* ln1b = (const float*)d_in[4];
  const float* w1   = (const float*)d_in[5];
  const float* w2   = (const float*)d_in[6];
  const float* w3   = (const float*)d_in[7];
  const float* ln2w = (const float*)d_in[8];
  const float* ln2b = (const float*)d_in[9];
  const float* clw  = (const float*)d_in[10];
  const float* clb  = (const float*)d_in[11];
  const float* capu = (const float*)d_in[12];
  const float* cap  = (const float*)d_in[13];
  float* out = (float*)d_out;

  char* ws = (char*)d_ws;
  _Float16* xhat = (_Float16*)ws;  ws += (size_t)N_ * D_   * sizeof(_Float16);
  _Float16* xn   = (_Float16*)ws;  ws += (size_t)N_ * D_   * sizeof(_Float16);
  _Float16* Hbuf = (_Float16*)ws;  ws += (size_t)N_ * DFF  * sizeof(_Float16);
  float* Ybuf    = (float*)ws;     ws += (size_t)N_ * D_   * sizeof(float);
  float* coef    = (float*)ws;     ws += (size_t)N_ * E_   * sizeof(float);
  float* cmb     = (float*)ws;     ws += (size_t)N_ * D_   * sizeof(float);
  _Float16* w1t  = (_Float16*)ws;  ws += (size_t)D_ * DFF  * sizeof(_Float16);
  _Float16* w2t  = (_Float16*)ws;  ws += (size_t)D_ * DFF  * sizeof(_Float16);
  _Float16* w3t  = (_Float16*)ws;  ws += (size_t)D_ * DFF  * sizeof(_Float16);

  (void)hipMemsetAsync(cmb, 0, (size_t)N_ * D_ * sizeof(float), stream);
  prep_tokens<<<N_, 256, 0, stream>>>(x, ew, eidx, capu, cap, xhat, coef);

  dim3 blk(256);
  dim3 g12(DFF / BN, N_ / BM);     // 32 x 64
  dim3 g3 (D_  / BN, N_ / BM);     //  8 x 64
  dim3 gt12(DFF / 32, D_ / 32);    // transpose grids
  dim3 gt3 (D_ / 32, DFF / 32);

  for (int e = 0; e < E_; ++e) {
    const float* W1 = w1 + (size_t)e * D_ * DFF;
    const float* W2 = w2 + (size_t)e * D_ * DFF;
    const float* W3 = w3 + (size_t)e * DFF * D_;
    int act = e % 3;

    ln1_apply<<<N_ * D_ / 256, blk, 0, stream>>>(xhat, ln1w + (size_t)e * D_,
                                                 ln1b + (size_t)e * D_, xn);
    transpose_to_f16<<<gt12, blk, 0, stream>>>(W1, w1t, D_, DFF);
    if (act == 0) transpose_to_f16<<<gt12, blk, 0, stream>>>(W2, w2t, D_, DFF);
    transpose_to_f16<<<gt3, blk, 0, stream>>>(W3, w3t, DFF, D_);

    if (act == 0)
      moe_gemm12<0><<<g12, blk, 0, stream>>>(xn, w1t, w2t, Hbuf);
    else if (act == 1)
      moe_gemm12<1><<<g12, blk, 0, stream>>>(xn, w1t, w1t, Hbuf);
    else
      moe_gemm12<2><<<g12, blk, 0, stream>>>(xn, w1t, w1t, Hbuf);

    moe_gemm3<<<g3, blk, 0, stream>>>(Hbuf, w3t, x, Ybuf);
    ln2_combine<<<N_, 256, 0, stream>>>(Ybuf, ln2w + (size_t)e * D_,
                                        ln2b + (size_t)e * D_, coef, e, cmb);
  }
  final_ln<<<N_, 256, 0, stream>>>(cmb, clw, clb, out);
}
